// GeminiAttention_76244259439334
// MI455X (gfx1250) — matrix operand order repaired and checked by reading, not compiled, not run on hardware
//
#include <hip/hip_runtime.h>
#include <cstdint>

#define DMODEL 1024
#define NHEADS 16
#define DKH    64
#define BB     2
#define SS     2048
#define MROWS  (BB * SS)

typedef __attribute__((ext_vector_type(16))) _Float16 v16h;
typedef __attribute__((ext_vector_type(8)))  float    v8f;

union Frag16 { v16h v; uint32_t u[8]; _Float16 h[16]; };

__device__ __forceinline__ uint32_t ldpair(const _Float16* p) {
  return *(const uint32_t*)p;
}

__device__ __forceinline__ v8f wmma_f16(v16h a, v16h b, v8f c) {
  // D = A(16x32 f16) * B(32x16 f16) + C(16x16 f32)
  return __builtin_amdgcn_wmma_f32_16x16x32_f16(false, a, false, b, (short)0, c,
                                                false, false);
}

// ---------------------------------------------------------------- convert
__global__ void cvt_f32_to_f16(const float* __restrict__ in,
                               _Float16* __restrict__ out, int n) {
  int i = blockIdx.x * blockDim.x + threadIdx.x;
  int stride = gridDim.x * blockDim.x;
  for (; i < n; i += stride) out[i] = (_Float16)in[i];
}

// ---------------------------------------------------------------- GEMM
// C[M,N] = (A[M,K] * B[K,N] + bias[N]) * cscale
// A,B f16 row-major; M%256==0, N%64==0, K%32==0.
// BM=256, BN=64, BK=32; 8 waves as 4(M)x2(N); wave tile 64x32 = 8 WMMA frags.
// Software pipeline: stage next K-tile in registers while computing current.
template <typename OutT>
__global__ void __launch_bounds__(256) gemm_wmma_k(
    const _Float16* __restrict__ A, const _Float16* __restrict__ B,
    const float* __restrict__ bias, OutT* __restrict__ C,
    int M, int N, int K, float cscale) {
  constexpr int BM = 256, BN = 64, BK = 32, LDA = 40, LDB = 40;
  __shared__ __align__(16) _Float16 As[BM][LDA];
  __shared__ __align__(16) _Float16 Bs[BN][LDB];   // transposed: [n][k]

  const int tid  = threadIdx.x;
  const int lane = tid & 31, wid = tid >> 5;
  const int warpM = wid >> 1, warpN = wid & 1;     // 4 x 2 waves
  const int r16 = lane & 15, khf = lane >> 4;
  const int blockM = blockIdx.y * BM, blockN = blockIdx.x * BN;

  // staging assignment: thread tid owns full A row `tid` (32 halves = 4x uint4)
  const int kB  = tid >> 3;            // 0..31
  const int n0B = (tid & 7) * 8;       // 0..56

  const _Float16* gA = A + (size_t)(blockM + tid) * K;
  const _Float16* gB = B + (size_t)kB * N + blockN + n0B;

  uint4 ra[4], rb;
  auto load_tile = [&](int k0) {
    const uint4* pa = (const uint4*)(gA + k0);
#pragma unroll
    for (int i = 0; i < 4; ++i) ra[i] = pa[i];
    rb = *(const uint4*)(gB + (size_t)k0 * N);
  };
  auto store_tile = [&]() {
#pragma unroll
    for (int i = 0; i < 4; ++i) *(uint4*)&As[tid][i * 8] = ra[i];
    const _Float16* bh = (const _Float16*)&rb;
#pragma unroll
    for (int j = 0; j < 8; ++j) Bs[n0B + j][kB] = bh[j];
  };

  v8f acc[4][2] = {};

  load_tile(0);
  store_tile();
  __syncthreads();

  for (int k0 = 0;;) {
    const int knext = k0 + BK;
    const bool more = knext < K;
    if (more) {
      load_tile(knext);
      __builtin_prefetch(gA + knext + BK, 0, 0);
    }

    Frag16 af[4], bf[2];
#pragma unroll
    for (int fm = 0; fm < 4; ++fm) {
      const _Float16* base = &As[warpM * 64 + fm * 16 + r16][0];
#pragma unroll
      for (int i = 0; i < 8; ++i) {
        int kk = 2 * (i & 3) + 16 * (i >> 2) + 8 * khf;   // A-frag 16-bit layout
        af[fm].u[i] = ldpair(base + kk);
      }
    }
#pragma unroll
    for (int fn = 0; fn < 2; ++fn) {
      const _Float16* base = &Bs[warpN * 32 + fn * 16 + r16][0];
#pragma unroll
      for (int i = 0; i < 8; ++i) {
        int kk = 2 * (i & 3) + 16 * (i >> 2) + 8 * khf;   // B-frag 16-bit layout
        bf[fn].u[i] = ldpair(base + kk);
      }
    }
#pragma unroll
    for (int fm = 0; fm < 4; ++fm)
#pragma unroll
      for (int fn = 0; fn < 2; ++fn)
        acc[fm][fn] = wmma_f16(af[fm].v, bf[fn].v, acc[fm][fn]);

    if (!more) break;
    __syncthreads();   // everyone done reading current tiles
    store_tile();
    __syncthreads();   // next tiles visible
    k0 = knext;
  }

#pragma unroll
  for (int fm = 0; fm < 4; ++fm)
#pragma unroll
    for (int fn = 0; fn < 2; ++fn)
#pragma unroll
      for (int r = 0; r < 8; ++r) {
        int gm = blockM + warpM * 64 + fm * 16 + r + 8 * khf;
        int gn = blockN + warpN * 32 + fn * 16 + r16;
        float v = (acc[fm][fn][r] + bias[gn]) * cscale;
        C[(size_t)gm * N + gn] = (OutT)v;
      }
}

// ---------------------------------------------------------------- attention
// One wave per (b, h, 16-query-row block). MQA: K/V shared across heads.
// Q is pre-scaled by 1/sqrt(dk) in the projection GEMM epilogue.
template <bool MASKED>
__device__ __forceinline__ void attn_chunk(
    int t0, int q0, int b, int r16, int khf,
    const Frag16 (&qf)[2], const _Float16* __restrict__ Kh,
    const _Float16* __restrict__ V,
    v8f (&o)[4], float (&mrow)[8], float (&lrow)[8],
    _Float16 (*P)[40]) {
  v8f s[2];
#pragma unroll
  for (int tb = 0; tb < 2; ++tb) {
    Frag16 kf[2];
    const int t = t0 + tb * 16 + r16;
    const _Float16* kbase = Kh + ((size_t)(b * SS) + t) * DKH;
#pragma unroll
    for (int j = 0; j < 2; ++j)
#pragma unroll
      for (int i = 0; i < 8; ++i) {
        int kk = 2 * (i & 3) + 16 * (i >> 2) + 8 * khf + 32 * j;
        kf[j].u[i] = ldpair(kbase + kk);
      }
    v8f sv = {};
    sv = wmma_f16(qf[0].v, kf[0].v, sv);
    sv = wmma_f16(qf[1].v, kf[1].v, sv);
    s[tb] = sv;
  }
  if (MASKED) {
#pragma unroll
    for (int tb = 0; tb < 2; ++tb)
#pragma unroll
      for (int r = 0; r < 8; ++r) {
        int qrow = q0 + r + 8 * khf;
        int tcol = t0 + tb * 16 + r16;
        if (tcol > qrow) s[tb][r] = -1.0e9f;
      }
  }
  // online softmax: rows live across 16-lane groups -> shfl_xor reductions
#pragma unroll
  for (int r = 0; r < 8; ++r) {
    float cmax = fmaxf(s[0][r], s[1][r]);
#pragma unroll
    for (int off = 8; off >= 1; off >>= 1)
      cmax = fmaxf(cmax, __shfl_xor(cmax, off, 32));
    float mnew = fmaxf(mrow[r], cmax);
    float corr = __expf(mrow[r] - mnew);
    float p0 = __expf(s[0][r] - mnew);
    float p1 = __expf(s[1][r] - mnew);
    float rs = p0 + p1;
#pragma unroll
    for (int off = 8; off >= 1; off >>= 1)
      rs += __shfl_xor(rs, off, 32);
    lrow[r] = lrow[r] * corr + rs;
    mrow[r] = mnew;
#pragma unroll
    for (int j = 0; j < 4; ++j) o[j][r] *= corr;
    s[0][r] = p0;
    s[1][r] = p1;
  }
  // P (C-frag layout) -> per-wave LDS tile -> reload in A-frag layout
#pragma unroll
  for (int tb = 0; tb < 2; ++tb)
#pragma unroll
    for (int r = 0; r < 8; ++r)
      P[r + 8 * khf][tb * 16 + r16] = (_Float16)s[tb][r];
  asm volatile("s_wait_dscnt 0" ::: "memory");
  Frag16 pf;
#pragma unroll
  for (int i = 0; i < 8; ++i) {
    int kk = 2 * (i & 3) + 16 * (i >> 2) + 8 * khf;
    pf.u[i] = ldpair(&P[r16][kk]);
  }
  // O(16x64) += P(16x32) * V(32x64), four n-blocks of 16
#pragma unroll
  for (int j = 0; j < 4; ++j) {
    Frag16 vf;
    const int c = j * 16 + r16;
#pragma unroll
    for (int vv = 0; vv < 8; ++vv) {
      int krow = 2 * (vv & 3) + 16 * (vv >> 2) + 8 * khf;
      vf.h[2 * vv]     = V[((size_t)(b * SS) + t0 + krow)     * DKH + c];
      vf.h[2 * vv + 1] = V[((size_t)(b * SS) + t0 + krow + 1) * DKH + c];
    }
    o[j] = wmma_f16(pf.v, vf.v, o[j]);
  }
}

__global__ void __launch_bounds__(256) attn_fa_k(
    const _Float16* __restrict__ Q, const _Float16* __restrict__ Kh,
    const _Float16* __restrict__ V, _Float16* __restrict__ O) {
  __shared__ __align__(16) _Float16 Plds[8][16][40];

  const int tid  = threadIdx.x;
  const int lane = tid & 31, wid = tid >> 5;
  const int r16 = lane & 15, khf = lane >> 4;
  const int gw = blockIdx.x * 8 + wid;
  const int qb = gw & 127;           // S/16 = 128 query blocks
  const int h  = (gw >> 7) & 15;
  const int b  = gw >> 11;
  const int q0 = qb * 16;

  // Q fragments: two 16x32 frags covering dk 0..31 and 32..63
  Frag16 qf[2];
  const _Float16* qbase = Q + ((size_t)(b * SS) + q0 + r16) * DMODEL + h * DKH;
#pragma unroll
  for (int j = 0; j < 2; ++j)
#pragma unroll
    for (int i = 0; i < 8; ++i) {
      int kk = 2 * (i & 3) + 16 * (i >> 2) + 8 * khf + 32 * j;
      qf[j].u[i] = ldpair(qbase + kk);
    }

  v8f o[4] = {};
  float mrow[8], lrow[8];
#pragma unroll
  for (int r = 0; r < 8; ++r) { mrow[r] = -3.0e38f; lrow[r] = 0.0f; }

  _Float16 (*P)[40] = Plds[wid];

  // full (mask-free) chunks: t0+31 <= q0, then masked diagonal chunks
  const int tFull = (q0 >= 31) ? ((((q0 - 31) >> 5) + 1) << 5) : 0;
  int t0 = 0;
  for (; t0 < tFull; t0 += 32)
    attn_chunk<false>(t0, q0, b, r16, khf, qf, Kh, V, o, mrow, lrow, P);
  for (; t0 < q0 + 16; t0 += 32)
    attn_chunk<true>(t0, q0, b, r16, khf, qf, Kh, V, o, mrow, lrow, P);

#pragma unroll
  for (int j = 0; j < 4; ++j)
#pragma unroll
    for (int r = 0; r < 8; ++r) {
      int m = q0 + r + 8 * khf;
      O[((size_t)(b * SS) + m) * DMODEL + h * DKH + j * 16 + r16] =
          (_Float16)(o[j][r] / lrow[r]);
    }
}

// ---------------------------------------------------------------- launch
extern "C" void kernel_launch(void* const* d_in, const int* in_sizes, int n_in,
                              void* d_out, int out_size, void* d_ws, size_t ws_size,
                              hipStream_t stream) {
  const float* x  = (const float*)d_in[0];
  // d_in[1] = mask (causal, computed analytically in-kernel)
  const float* wq = (const float*)d_in[2];
  const float* bq = (const float*)d_in[3];
  const float* wk = (const float*)d_in[4];
  const float* bk = (const float*)d_in[5];
  const float* wv = (const float*)d_in[6];
  const float* bv = (const float*)d_in[7];
  const float* wo = (const float*)d_in[8];
  const float* bo = (const float*)d_in[9];
  float* out = (float*)d_out;

  char* ws = (char*)d_ws;
  size_t off = 0;
  auto take = [&](size_t elems) {
    _Float16* p = (_Float16*)(ws + off);
    off = (off + elems * sizeof(_Float16) + 255) & ~(size_t)255;
    return p;
  };
  _Float16* x_h  = take((size_t)MROWS  * DMODEL);
  _Float16* wq_h = take((size_t)DMODEL * DMODEL);
  _Float16* wk_h = take((size_t)DMODEL * DKH);
  _Float16* wv_h = take((size_t)DMODEL * DKH);
  _Float16* wo_h = take((size_t)DMODEL * DMODEL);
  _Float16* q_h  = take((size_t)MROWS  * DMODEL);
  _Float16* k_h  = take((size_t)MROWS  * DKH);
  _Float16* v_h  = take((size_t)MROWS  * DKH);
  _Float16* ao_h = take((size_t)MROWS  * DMODEL);

  cvt_f32_to_f16<<<512, 256, 0, stream>>>(x,  x_h,  MROWS * DMODEL);
  cvt_f32_to_f16<<<256, 256, 0, stream>>>(wq, wq_h, DMODEL * DMODEL);
  cvt_f32_to_f16<<<64,  256, 0, stream>>>(wk, wk_h, DMODEL * DKH);
  cvt_f32_to_f16<<<64,  256, 0, stream>>>(wv, wv_h, DMODEL * DKH);
  cvt_f32_to_f16<<<256, 256, 0, stream>>>(wo, wo_h, DMODEL * DMODEL);

  const float qscale = 0.125f;   // 1/sqrt(DKH)
  dim3 blk(256);
  dim3 gQ(DMODEL / 64, MROWS / 256);   // (16, 16)
  dim3 gKV(DKH / 64, MROWS / 256);     // (1, 16)
  gemm_wmma_k<_Float16><<<gQ,  blk, 0, stream>>>(x_h, wq_h, bq, q_h, MROWS, DMODEL, DMODEL, qscale);
  gemm_wmma_k<_Float16><<<gKV, blk, 0, stream>>>(x_h, wk_h, bk, k_h, MROWS, DKH,    DMODEL, 1.0f);
  gemm_wmma_k<_Float16><<<gKV, blk, 0, stream>>>(x_h, wv_h, bv, v_h, MROWS, DKH,    DMODEL, 1.0f);

  attn_fa_k<<<(BB * NHEADS * (SS / 16)) / 8, blk, 0, stream>>>(q_h, k_h, v_h, ao_h);

  gemm_wmma_k<float><<<gQ, blk, 0, stream>>>(ao_h, wo_h, bo, out, MROWS, DMODEL, DMODEL, 1.0f);

  (void)in_sizes; (void)n_in; (void)out_size; (void)ws_size;
}
